// RGCN_66924180407122
// MI455X (gfx1250) — matrix operand order repaired and checked by reading, not compiled, not run on hardware
//
#include <hip/hip_runtime.h>

// ---------------------------------------------------------------------------
// RGCN 2-layer forward for MI455X (gfx1250, wave32, WMMA bf16 16x16x32).
// Transform-then-aggregate: per-relation 16-edge tiles feed v_wmma, outputs
// are mean-scaled and atomically accumulated into L2-resident buffers.
// ---------------------------------------------------------------------------

typedef __attribute__((ext_vector_type(16))) __bf16 v16bf;
typedef __attribute__((ext_vector_type(8)))  __bf16 v8bf;
typedef __attribute__((ext_vector_type(8)))  float  v8f;

#define NNODE 100000
#define NREL  16
#define NEDGE 1600000
#define DF    64

// A-operand loader for V_WMMA_F32_16X16X32_BF16.
// Per ISA layout (16-bit A 16x32): lane half selects K blocks; per lane the
// needed 16 bf16 are two contiguous 8-element (16B) runs:
//   a[0..7]  = row[ks*32 + 8*half + 0..7]
//   a[8..15] = row[ks*32 + 16 + 8*half + 0..7]
static __device__ __forceinline__ v16bf load_a(const __bf16* p) {
  const v8bf* q = (const v8bf*)p;
  v8bf lo = q[0];
  v8bf hi = q[2]; // +16 elements
  v16bf a;
#pragma unroll
  for (int j = 0; j < 8; ++j) { a[j] = lo[j]; a[j + 8] = hi[j]; }
  return a;
}

// ---------------------------------------------------------------------------
// Setup kernels
// ---------------------------------------------------------------------------

__global__ void k_hist(const int* __restrict__ et, const int* __restrict__ dst,
                       float* __restrict__ cnt, int* __restrict__ hist) {
  int e = blockIdx.x * blockDim.x + threadIdx.x;
  if (e >= NEDGE) return;
  int r = et[e], d = dst[e];
  atomicAdd(&cnt[r * NNODE + d], 1.0f);
  atomicAdd(&hist[r], 1);
}

__global__ void k_prefix(const int* __restrict__ hist, int* __restrict__ off,
                         int* __restrict__ ctr) {
  if (threadIdx.x == 0 && blockIdx.x == 0) {
    int a = 0;
    for (int r = 0; r < NREL; ++r) { off[r] = a; ctr[r] = a; a += hist[r]; }
    off[NREL] = a;
  }
}

__global__ void k_scatter(const int* __restrict__ et, int* __restrict__ ctr,
                          int* __restrict__ sorted) {
  int e = blockIdx.x * blockDim.x + threadIdx.x;
  if (e >= NEDGE) return;
  int pos = atomicAdd(&ctr[et[e]], 1);
  sorted[pos] = e;
}

__global__ void k_cvt(const float* __restrict__ x, __bf16* __restrict__ xb, int n) {
  int i = blockIdx.x * blockDim.x + threadIdx.x;
  if (i < n) xb[i] = (__bf16)x[i];
}

// Swizzle a [numR][64][ncols] f32 weight into the WMMA B-operand bf16 layout:
// chunk = (r*2+ks)*NT + nt ; within chunk: lane*16 + j ;
// element j of lane L -> K = 32*ks + 8*(L>>4) + (j&7) + 16*(j>>3), N = (L&15) + 16*nt.
__global__ void k_swizzleB(const float* __restrict__ src, __bf16* __restrict__ dst,
                           int NT, int ncols, int total) {
  int id = blockIdx.x * blockDim.x + threadIdx.x;
  if (id >= total) return;
  int j = id & 15;
  int lane = (id >> 4) & 31;
  int chunk = id >> 9;
  int nt = chunk % NT;
  int ks = (chunk / NT) & 1;
  int r  = chunk / (NT * 2);
  int k = 32 * ks + 8 * (lane >> 4) + (j & 7) + 16 * (j >> 3);
  int n = (lane & 15) + 16 * nt;
  dst[id] = (__bf16)src[(r * 64 + k) * ncols + n];
}

// ---------------------------------------------------------------------------
// Root GEMM: out[node] = A[node] @ rootB + bias. One wave per 16-node tile.
// NT = number of 16-wide output column tiles (4 -> 64 cols, 1 -> 16 cols).
// ---------------------------------------------------------------------------
template <int NT>
__global__ __launch_bounds__(32) void k_rootgemm(const __bf16* __restrict__ A,
                                                 const __bf16* __restrict__ Bsw,
                                                 const float* __restrict__ bias,
                                                 float* __restrict__ out) {
  constexpr int NC = NT * 16;
  int lane = threadIdx.x;
  int half = lane >> 4;
  int m = lane & 15;
  int node0 = blockIdx.x * 16;

  v8f acc[NT];
#pragma unroll
  for (int t = 0; t < NT; ++t) acc[t] = (v8f)0.0f;

  const __bf16* rowp = A + (node0 + m) * DF + half * 8;
#pragma unroll
  for (int ks = 0; ks < 2; ++ks) {
    v16bf a = load_a(rowp + ks * 32);
#pragma unroll
    for (int t = 0; t < NT; ++t) {
      v16bf b = *(const v16bf*)(Bsw + (ks * NT + t) * 512 + lane * 16);
      acc[t] = __builtin_amdgcn_wmma_f32_16x16x32_bf16(
          false, a, false, b, (short)0, acc[t], false, false);
    }
  }
#pragma unroll
  for (int t = 0; t < NT; ++t) {
    int col = m + 16 * t;
    float bv = bias[col];
#pragma unroll
    for (int i = 0; i < 8; ++i) {
      int row = node0 + i + 8 * half; // C layout: vgpr i, lane half -> M
      out[row * NC + col] = acc[t][i] + bv;
    }
  }
}

// ---------------------------------------------------------------------------
// Edge kernel: one wave per 16-edge tile inside one relation segment.
// y = gather(A, src) @ W[r]; each row scaled by 1/max(cnt[r,dst],1);
// atomically accumulated into out[dst]. Padded rows use scale 0 so EXEC can
// stay all-ones for the WMMAs.
// ---------------------------------------------------------------------------
template <int NT>
__global__ __launch_bounds__(32) void k_edges(const __bf16* __restrict__ A,
                                              const __bf16* __restrict__ Wsw,
                                              const int* __restrict__ srcIdx,
                                              const int* __restrict__ dstIdx,
                                              const int* __restrict__ sorted,
                                              const int* __restrict__ off,
                                              const float* __restrict__ cnt,
                                              float* __restrict__ out) {
  constexpr int NC = NT * 16;
  __shared__ int   lds_src[16];
  __shared__ int   lds_dst[16];
  __shared__ float lds_scl[16];

  int t = blockIdx.x;
  int lane = threadIdx.x;

  // Map global tile id -> (relation, edge range) using per-relation offsets.
  int r = -1, e0 = 0, segEnd = 0, base = 0;
  for (int rr = 0; rr < NREL; ++rr) {
    int o0 = off[rr], o1 = off[rr + 1];
    int ntiles = (o1 - o0 + 15) >> 4;
    if (t < base + ntiles) { r = rr; e0 = o0 + ((t - base) << 4); segEnd = o1; break; }
    base += ntiles;
  }
  if (r < 0) return;

  if (lane < 16) {
    int e = e0 + lane;
    bool valid = e < segEnd;
    int es = sorted[valid ? e : (segEnd - 1)];
    int s = srcIdx[es];
    int d = dstIdx[es];
    lds_src[lane] = s;
    lds_dst[lane] = d;
    float c = cnt[r * NNODE + d];
    lds_scl[lane] = valid ? (1.0f / fmaxf(c, 1.0f)) : 0.0f;
  }
  __syncthreads();

  int half = lane >> 4;
  int m = lane & 15;
  const __bf16* rowp = A + lds_src[m] * DF + half * 8;

  v8f acc[NT];
#pragma unroll
  for (int tt = 0; tt < NT; ++tt) acc[tt] = (v8f)0.0f;

#pragma unroll
  for (int ks = 0; ks < 2; ++ks) {
    v16bf a = load_a(rowp + ks * 32);
#pragma unroll
    for (int tt = 0; tt < NT; ++tt) {
      v16bf b = *(const v16bf*)(Wsw + ((r * 2 + ks) * NT + tt) * 512 + lane * 16);
      acc[tt] = __builtin_amdgcn_wmma_f32_16x16x32_bf16(
          false, a, false, b, (short)0, acc[tt], false, false);
    }
  }

#pragma unroll
  for (int i = 0; i < 8; ++i) {
    int row = i + 8 * half;
    float scl = lds_scl[row];
    int d = lds_dst[row];
#pragma unroll
    for (int tt = 0; tt < NT; ++tt) {
      int col = m + 16 * tt;
      atomicAdd(&out[d * NC + col], acc[tt][i] * scl);
    }
  }
}

// ---------------------------------------------------------------------------
// Elementwise: relu(h) in place + bf16 copy for next layer's WMMA A operand.
// ---------------------------------------------------------------------------
__global__ void k_relucvt(float* __restrict__ h, __bf16* __restrict__ hb, int n) {
  int i = blockIdx.x * blockDim.x + threadIdx.x;
  if (i < n) {
    float v = fmaxf(h[i], 0.0f);
    h[i] = v;
    hb[i] = (__bf16)v;
  }
}

// Final relu + log_softmax over 16 classes, one thread per node.
__global__ void k_lsm(float* __restrict__ out) {
  int n = blockIdx.x * blockDim.x + threadIdx.x;
  if (n >= NNODE) return;
  float v[16];
  float4* p = (float4*)(out + n * 16);
#pragma unroll
  for (int q = 0; q < 4; ++q) {
    float4 tv = p[q];
    v[4 * q + 0] = fmaxf(tv.x, 0.0f);
    v[4 * q + 1] = fmaxf(tv.y, 0.0f);
    v[4 * q + 2] = fmaxf(tv.z, 0.0f);
    v[4 * q + 3] = fmaxf(tv.w, 0.0f);
  }
  float mx = v[0];
#pragma unroll
  for (int q = 1; q < 16; ++q) mx = fmaxf(mx, v[q]);
  float s = 0.0f;
#pragma unroll
  for (int q = 0; q < 16; ++q) s += __expf(v[q] - mx);
  float lse = mx + __logf(s);
#pragma unroll
  for (int q = 0; q < 4; ++q) {
    float4 tv;
    tv.x = v[4 * q + 0] - lse;
    tv.y = v[4 * q + 1] - lse;
    tv.z = v[4 * q + 2] - lse;
    tv.w = v[4 * q + 3] - lse;
    p[q] = tv;
  }
}

// ---------------------------------------------------------------------------
// Host launcher
// ---------------------------------------------------------------------------
extern "C" void kernel_launch(void* const* d_in, const int* in_sizes, int n_in,
                              void* d_out, int out_size, void* d_ws, size_t ws_size,
                              hipStream_t stream) {
  (void)in_sizes; (void)n_in; (void)out_size; (void)ws_size;

  const float* x     = (const float*)d_in[0];
  const int*   eidx  = (const int*)d_in[1];    // (2, E): src then dst
  const int*   etyp  = (const int*)d_in[2];
  const float* W1    = (const float*)d_in[3];  // (16, 64, 64)
  const float* root1 = (const float*)d_in[4];  // (64, 64)
  const float* b1    = (const float*)d_in[5];  // (64,)
  const float* W2    = (const float*)d_in[6];  // (16, 64, 16)
  const float* root2 = (const float*)d_in[7];  // (64, 16)
  const float* b2    = (const float*)d_in[8];  // (16,)
  float* out = (float*)d_out;                  // (N, 16)

  const int* srcIdx = eidx;
  const int* dstIdx = eidx + NEDGE;

  char* ws = (char*)d_ws;
  // Workspace layout (256B-aligned blocks)
  float*  cnt    = (float*)(ws + 0);                       //  6,400,000 B
  int*    hist   = (int*)(ws + 6400000);                   //         64 B
  int*    off    = (int*)(ws + 6400128);                   //         68 B
  int*    ctr    = (int*)(ws + 6400256);                   //         64 B
  int*    sorted = (int*)(ws + 6400512);                   //  6,400,000 B
  __bf16* xb     = (__bf16*)(ws + 12800512);               // 12,800,000 B
  float*  h1     = (float*)(ws + 25600512);                // 25,600,000 B
  __bf16* h1b    = (__bf16*)(ws + 51200512);               // 12,800,000 B
  __bf16* w1s    = (__bf16*)(ws + 64000512);               //    131,072 B
  __bf16* w2s    = (__bf16*)(ws + 64131584);               //     32,768 B
  __bf16* r1s    = (__bf16*)(ws + 64164352);               //     16,384 B
  __bf16* r2s    = (__bf16*)(ws + 64180736);               //      4,096 B

  const int edgeBlocks = (NEDGE + 255) / 256;          // 6250
  const int featBlocks = (NNODE * DF + 255) / 256;     // 25000
  const int nodeTiles  = NNODE / 16;                   // 6250
  const int edgeTiles  = NEDGE / 16 + NREL;            // 100016

  // 0) zero cnt + hist (off/ctr are fully written by k_prefix)
  hipMemsetAsync(cnt, 0, (size_t)NREL * NNODE * sizeof(float) + 64, stream);

  // 1) counts per (relation, dst) + per-relation histogram
  k_hist<<<edgeBlocks, 256, 0, stream>>>(etyp, dstIdx, cnt, hist);

  // 2) 16-bucket prefix sum, 3) counting-sort edge ids by relation
  k_prefix<<<1, 32, 0, stream>>>(hist, off, ctr);
  k_scatter<<<edgeBlocks, 256, 0, stream>>>(etyp, ctr, sorted);

  // 4) bf16 feature copy + B-operand swizzles of all weight matrices
  k_cvt<<<featBlocks, 256, 0, stream>>>(x, xb, NNODE * DF);
  k_swizzleB<<<(65536 + 255) / 256, 256, 0, stream>>>(W1, w1s, 4, 64, 65536);
  k_swizzleB<<<(16384 + 255) / 256, 256, 0, stream>>>(W2, w2s, 1, 16, 16384);
  k_swizzleB<<<(4096 + 255) / 256, 256, 0, stream>>>(root1, r1s, 4, 64, 4096);
  k_swizzleB<<<(1024 + 255) / 256, 256, 0, stream>>>(root2, r2s, 1, 16, 1024);

  // ---- Layer 1: h1 = x @ root1 + b1  (init) ; += mean-aggregated x@W1[r]
  k_rootgemm<4><<<nodeTiles, 32, 0, stream>>>(xb, r1s, b1, h1);
  k_edges<4><<<edgeTiles, 32, 0, stream>>>(xb, w1s, srcIdx, dstIdx, sorted, off, cnt, h1);

  // relu(relu(.)) == relu(.), plus bf16 copy for layer-2 A operand
  k_relucvt<<<featBlocks, 256, 0, stream>>>(h1, h1b, NNODE * DF);

  // ---- Layer 2: out = h1 @ root2 + b2 ; += mean-aggregated h1@W2[r]
  k_rootgemm<1><<<nodeTiles, 32, 0, stream>>>(h1b, r2s, b2, out);
  k_edges<1><<<edgeTiles, 32, 0, stream>>>(h1b, w2s, srcIdx, dstIdx, sorted, off, cnt, out);

  // final relu + log_softmax over 16 classes
  k_lsm<<<(NNODE + 255) / 256, 256, 0, stream>>>(out);
}
